// MoE_68075231641816
// MI455X (gfx1250) — matrix operand order repaired and checked by reading, not compiled, not run on hardware
//
#include <hip/hip_runtime.h>
#include <hip/hip_bf16.h>
#include <stdint.h>

// ---------------------------------------------------------------------------
// MoE (SwiGLU experts, top-2 of 8 + shared expert) for MI455X / gfx1250.
// Compute-bound at bf16 WMMA => convert fp32 params/acts to bf16 once,
// run all GEMMs through v_wmma_f32_16x16x32_bf16, gather routed tokens per
// expert. Global->LDS staging uses CDNA5 async-to-LDS (ASYNCcnt) with
// double-buffered tiles: issue batch k+1, s_wait_asynccnt(4) to drain only
// batch k, compute while k+1 streams in.
// ---------------------------------------------------------------------------

#define T_TOK 8192   // 4 * 2048 tokens
#define DH    2048   // hidden dim
#define DE    1024   // expert ffn dim
#define SDIM  2048   // shared expert ffn dim (D_EXPERT * N_SHARED)
#define NEXP  8

typedef __attribute__((ext_vector_type(16))) __bf16 v16bf;
typedef __attribute__((ext_vector_type(8)))  float  v8f;
typedef __attribute__((ext_vector_type(4)))  __bf16 v4bf;
typedef int v4i __attribute__((vector_size(16)));

#define AS1 __attribute__((address_space(1)))
#define AS3 __attribute__((address_space(3)))

union FragAB { v16bf v; uint4 q[2]; };

__device__ __forceinline__ v8f v8f_zero() {
  v8f z;
#pragma unroll
  for (int i = 0; i < 8; ++i) z[i] = 0.f;
  return z;
}

// 16-byte global -> LDS copy. CDNA5 async-to-LDS path (ASYNCcnt).
__device__ __forceinline__ void async_cp16(const void* g, void* l) {
#if defined(__gfx1250__) && __has_builtin(__builtin_amdgcn_global_load_async_to_lds_b128)
  __builtin_amdgcn_global_load_async_to_lds_b128((AS1 v4i*)g, (AS3 v4i*)l, 0, 0);
#else
  *(uint4*)l = *(const uint4*)g;
#endif
}

// Drain all but the newest batch of 4 async loads (double-buffer steady state).
__device__ __forceinline__ void wait_async4() {
#if defined(__gfx1250__) && __has_builtin(__builtin_amdgcn_s_wait_asynccnt)
  __builtin_amdgcn_s_wait_asynccnt(4);
#endif
  __syncthreads();
}

// Drain everything (last iteration).
__device__ __forceinline__ void wait_async0() {
#if defined(__gfx1250__) && __has_builtin(__builtin_amdgcn_s_wait_asynccnt)
  __builtin_amdgcn_s_wait_asynccnt(0);
#endif
  __syncthreads();
}

__device__ __forceinline__ v8f wmma_bf16(v16bf a, v16bf b, v8f c) {
#if defined(__gfx1250__)
  // 8 args: (neg_a, A, neg_b, B, c_mod, C, reuse_a, reuse_b)
  return __builtin_amdgcn_wmma_f32_16x16x32_bf16(false, a, false, b, (short)0, c,
                                                 false, false);
#else
  (void)a; (void)b;
  return c;
#endif
}

__device__ __forceinline__ float silu_f(float g) {
  return g * (1.f / (1.f + __expf(-g)));
}

// ---------------------------------------------------------------------------
// fp32 -> bf16 conversion (vectorized, grid-stride)
// ---------------------------------------------------------------------------
__global__ void __launch_bounds__(256)
k_cvt(const float* __restrict__ s, __bf16* __restrict__ d, long n) {
  long i = ((long)blockIdx.x * 256 + threadIdx.x) * 4;
  long stride = (long)gridDim.x * 256 * 4;
  for (; i < n; i += stride) {
    float4 f = *(const float4*)(s + i);
    v4bf o;
    o.x = (__bf16)f.x; o.y = (__bf16)f.y; o.z = (__bf16)f.z; o.w = (__bf16)f.w;
    *(v4bf*)(d + i) = o;
  }
}

__global__ void k_zero_counts(int* __restrict__ c) {
  if (threadIdx.x < NEXP) c[threadIdx.x] = 0;
}

// ---------------------------------------------------------------------------
// Router: one wave32 per token. scores = softmax(x @ W_g); unsorted top-2;
// scatter (token, prob) into per-expert lists.
// ---------------------------------------------------------------------------
__global__ void __launch_bounds__(256)
k_router(const float* __restrict__ x, const float* __restrict__ Wg,
         int* __restrict__ counts, int* __restrict__ idxl,
         float* __restrict__ wgtl) {
  int lane = threadIdx.x & 31;
  int wave = threadIdx.x >> 5;
  int t = blockIdx.x * 8 + wave;
  if (t >= T_TOK) return;

  float acc[NEXP];
#pragma unroll
  for (int e = 0; e < NEXP; ++e) acc[e] = 0.f;
  const float* xr = x + (size_t)t * DH;
  for (int d = lane; d < DH; d += 32) {
    float xv = xr[d];
    const float* wr = Wg + (size_t)d * NEXP;
#pragma unroll
    for (int e = 0; e < NEXP; ++e) acc[e] += xv * wr[e];
  }
#pragma unroll
  for (int e = 0; e < NEXP; ++e) {
    acc[e] += __shfl_xor(acc[e], 16, 32);
    acc[e] += __shfl_xor(acc[e], 8, 32);
    acc[e] += __shfl_xor(acc[e], 4, 32);
    acc[e] += __shfl_xor(acc[e], 2, 32);
    acc[e] += __shfl_xor(acc[e], 1, 32);
  }
  if (lane == 0) {
    float mx = acc[0];
#pragma unroll
    for (int e = 1; e < NEXP; ++e) mx = fmaxf(mx, acc[e]);
    float p[NEXP], s = 0.f;
#pragma unroll
    for (int e = 0; e < NEXP; ++e) { p[e] = __expf(acc[e] - mx); s += p[e]; }
    float inv = 1.f / s;
#pragma unroll
    for (int e = 0; e < NEXP; ++e) p[e] *= inv;
    int i0 = 0;
#pragma unroll
    for (int e = 1; e < NEXP; ++e) if (p[e] > p[i0]) i0 = e;
    int i1 = (i0 == 0) ? 1 : 0;
#pragma unroll
    for (int e = 0; e < NEXP; ++e) if (e != i0 && p[e] > p[i1]) i1 = e;
    int pos0 = atomicAdd(&counts[i0], 1);
    idxl[i0 * T_TOK + pos0] = t;
    wgtl[i0 * T_TOK + pos0] = p[i0];
    int pos1 = atomicAdd(&counts[i1], 1);
    idxl[i1 * T_TOK + pos1] = t;
    wgtl[i1 * T_TOK + pos1] = p[i1];
  }
}

__global__ void k_scan(const int* __restrict__ counts, int* __restrict__ offs) {
  if (threadIdx.x == 0 && blockIdx.x == 0) {
    int o = 0;
    for (int e = 0; e < NEXP; ++e) { offs[e] = o; o += counts[e]; }
    offs[NEXP] = o;
  }
}

// ---------------------------------------------------------------------------
// SwiGLU up-projection GEMM:  H = silu(A @ Wg) * (A @ Wu)    (bf16 out)
// Tiles: BM=128 x BN=64 x BK=32, 256 threads = 8 waves (4 x 2),
// double-buffered async LDS staging.
// counts/offs/idxl == nullptr -> shared expert (identity rows, cnt = T_TOK).
// ---------------------------------------------------------------------------
__global__ void __launch_bounds__(256)
k_swiglu(const __bf16* __restrict__ X, const __bf16* __restrict__ Wg_all,
         const __bf16* __restrict__ Wu_all, size_t w_stride, int Hcols,
         __bf16* __restrict__ Hout, const int* __restrict__ counts,
         const int* __restrict__ offs, const int* __restrict__ idxl) {
  const int BM = 128, BN = 64, BK = 32;
  __shared__ __bf16 As[2][BM * BK];
  __shared__ __bf16 Bg[2][BK * BN];
  __shared__ __bf16 Bu[2][BK * BN];

  int tid = threadIdx.x, lane = tid & 31, wid = tid >> 5;
  int wm = wid & 3, wn = wid >> 2;  // 4 wave-rows x 2 wave-cols
  int e = blockIdx.z;
  int cnt = counts ? counts[e] : T_TOK;
  int row0 = blockIdx.x * BM;
  if (row0 >= cnt) return;
  int col0 = blockIdx.y * BN;

  const __bf16* Wg = Wg_all + (size_t)e * w_stride;
  const __bf16* Wu = Wu_all + (size_t)e * w_stride;
  const int* idx = idxl ? (idxl + e * T_TOK) : nullptr;
  __bf16* H = Hout + (offs ? ((size_t)offs[e] * Hcols) : (size_t)0);

  // per-thread A staging rows (fixed across k-loop)
  int rA0 = tid >> 2, rA1 = rA0 + 64;
  int offA = (tid & 3) * 8;
  long tok0, tok1;
  if (idx) {
    tok0 = (row0 + rA0 < cnt) ? idx[row0 + rA0] : 0;
    tok1 = (row0 + rA1 < cnt) ? idx[row0 + rA1] : 0;
  } else {
    tok0 = row0 + rA0;
    tok1 = row0 + rA1;
  }
  int rB = tid >> 3, offB = (tid & 7) * 8;

  // 4 async b128 issues per thread per tile batch
  auto load_tiles = [&](int b, int k0) {
    async_cp16(X + (size_t)tok0 * DH + k0 + offA, &As[b][rA0 * BK + offA]);
    async_cp16(X + (size_t)tok1 * DH + k0 + offA, &As[b][rA1 * BK + offA]);
    async_cp16(Wg + (size_t)(k0 + rB) * Hcols + col0 + offB, &Bg[b][rB * BN + offB]);
    async_cp16(Wu + (size_t)(k0 + rB) * Hcols + col0 + offB, &Bu[b][rB * BN + offB]);
  };

  v8f accg[2][2], accu[2][2];
#pragma unroll
  for (int i = 0; i < 2; ++i)
#pragma unroll
    for (int j = 0; j < 2; ++j) { accg[i][j] = v8f_zero(); accu[i][j] = v8f_zero(); }

  int m16 = lane & 15, kh = lane >> 4;

  load_tiles(0, 0);
  int buf = 0;
  for (int k0 = 0; k0 < DH; k0 += BK) {
    int kn = k0 + BK;
    if (kn < DH) {
      load_tiles(buf ^ 1, kn);               // stream next tile into other buffer
      if (kn + BK < DH)
        __builtin_prefetch(Wg + (size_t)(kn + BK + rB) * Hcols + col0 + offB, 0, 2);
      wait_async4();                         // drain only the current tile's batch
    } else {
      wait_async0();
    }

    FragAB a[2], bg[2], bu[2];
#pragma unroll
    for (int i = 0; i < 2; ++i) {
      // 16-bit A 16x32 layout: lane m (0..15), K halves of 8 selected by lane>>4
      const __bf16* pa = &As[buf][(wm * 32 + i * 16 + m16) * BK + kh * 8];
      a[i].q[0] = *(const uint4*)pa;
      a[i].q[1] = *(const uint4*)(pa + 16);
    }
#pragma unroll
    for (int j = 0; j < 2; ++j) {
      // B 32x16 layout: N column = lane&15, K in table order matching A
      const __bf16* pg = &Bg[buf][wn * 32 + j * 16 + m16];
      const __bf16* pu = &Bu[buf][wn * 32 + j * 16 + m16];
#pragma unroll
      for (int s = 0; s < 16; ++s) {
        int kk = (s < 8) ? (kh * 8 + s) : (16 + kh * 8 + (s - 8));
        bg[j].v[s] = pg[kk * BN];
        bu[j].v[s] = pu[kk * BN];
      }
    }
#pragma unroll
    for (int i = 0; i < 2; ++i)
#pragma unroll
      for (int j = 0; j < 2; ++j) {
        accg[i][j] = wmma_bf16(a[i].v, bg[j].v, accg[i][j]);
        accu[i][j] = wmma_bf16(a[i].v, bu[j].v, accu[i][j]);
      }
    __syncthreads();  // all reads of buf done before it is refilled next iter
    buf ^= 1;
  }

  // fused SwiGLU epilogue, C layout: m = r + 8*(lane>>4), n = lane&15
#pragma unroll
  for (int i = 0; i < 2; ++i)
#pragma unroll
    for (int j = 0; j < 2; ++j)
#pragma unroll
      for (int r = 0; r < 8; ++r) {
        int mi = row0 + wm * 32 + i * 16 + r + 8 * kh;
        int n = col0 + wn * 32 + j * 16 + m16;
        if (mi < cnt) {
          float g = accg[i][j][r];
          float u = accu[i][j][r];
          H[(size_t)mi * Hcols + n] = (__bf16)(silu_f(g) * u);
        }
      }
}

// ---------------------------------------------------------------------------
// Down-projection GEMM: Out(+)= (H @ Wd).  Tiles 128x128x32, 8 waves (2 x 4),
// double-buffered async LDS staging.
// idxl == nullptr -> plain store (shared expert); else scaled atomic scatter.
// ---------------------------------------------------------------------------
__global__ void __launch_bounds__(256)
k_down(const __bf16* __restrict__ Hin, const __bf16* __restrict__ Wd_all,
       size_t w_stride, int K, float* __restrict__ Out,
       const int* __restrict__ counts, const int* __restrict__ offs,
       const int* __restrict__ idxl, const float* __restrict__ wgtl) {
  const int BM = 128, BN = 128, BK = 32;
  __shared__ __bf16 As[2][BM * BK];
  __shared__ __bf16 Bs[2][BK * BN];

  int tid = threadIdx.x, lane = tid & 31, wid = tid >> 5;
  int wm = wid & 1, wn = wid >> 1;  // 2 wave-rows x 4 wave-cols
  int e = blockIdx.z;
  int cnt = counts ? counts[e] : T_TOK;
  int row0 = blockIdx.x * BM;
  if (row0 >= cnt) return;
  int col0 = blockIdx.y * BN;

  const __bf16* A = Hin + (offs ? ((size_t)offs[e] * K) : (size_t)0);
  const __bf16* B = Wd_all + (size_t)e * w_stride;

  v8f acc[4][2];
#pragma unroll
  for (int i = 0; i < 4; ++i)
#pragma unroll
    for (int j = 0; j < 2; ++j) acc[i][j] = v8f_zero();

  int rA0 = tid >> 2, rA1 = rA0 + 64, offA = (tid & 3) * 8;
  int rB0 = tid >> 4, rB1 = rB0 + 16, offB = (tid & 15) * 8;
  int m16 = lane & 15, kh = lane >> 4;

  auto load_tiles = [&](int b, int k0) {
    async_cp16(A + (size_t)(row0 + rA0) * K + k0 + offA, &As[b][rA0 * BK + offA]);
    async_cp16(A + (size_t)(row0 + rA1) * K + k0 + offA, &As[b][rA1 * BK + offA]);
    async_cp16(B + (size_t)(k0 + rB0) * DH + col0 + offB, &Bs[b][rB0 * BN + offB]);
    async_cp16(B + (size_t)(k0 + rB1) * DH + col0 + offB, &Bs[b][rB1 * BN + offB]);
  };

  load_tiles(0, 0);
  int buf = 0;
  for (int k0 = 0; k0 < K; k0 += BK) {
    int kn = k0 + BK;
    if (kn < K) {
      load_tiles(buf ^ 1, kn);
      if (kn + BK < K)
        __builtin_prefetch(B + (size_t)(kn + BK + rB0) * DH + col0 + offB, 0, 2);
      wait_async4();
    } else {
      wait_async0();
    }

    FragAB a[4], b[2];
#pragma unroll
    for (int i = 0; i < 4; ++i) {
      const __bf16* pa = &As[buf][(wm * 64 + i * 16 + m16) * BK + kh * 8];
      a[i].q[0] = *(const uint4*)pa;
      a[i].q[1] = *(const uint4*)(pa + 16);
    }
#pragma unroll
    for (int j = 0; j < 2; ++j) {
      const __bf16* pb = &Bs[buf][wn * 32 + j * 16 + m16];
#pragma unroll
      for (int s = 0; s < 16; ++s) {
        int kk = (s < 8) ? (kh * 8 + s) : (16 + kh * 8 + (s - 8));
        b[j].v[s] = pb[kk * BN];
      }
    }
#pragma unroll
    for (int i = 0; i < 4; ++i)
#pragma unroll
      for (int j = 0; j < 2; ++j) acc[i][j] = wmma_bf16(a[i].v, b[j].v, acc[i][j]);
    __syncthreads();
    buf ^= 1;
  }

#pragma unroll
  for (int i = 0; i < 4; ++i)
#pragma unroll
    for (int j = 0; j < 2; ++j)
#pragma unroll
      for (int r = 0; r < 8; ++r) {
        int gl = row0 + wm * 64 + i * 16 + r + 8 * kh;
        int n = col0 + wn * 32 + j * 16 + m16;
        if (gl < cnt) {
          if (idxl) {
            int tok = idxl[e * T_TOK + gl];
            float w = wgtl[e * T_TOK + gl];
            atomicAdd(&Out[(size_t)tok * DH + n], w * acc[i][j][r]);
          } else {
            Out[(size_t)gl * DH + n] = acc[i][j][r];
          }
        }
      }
}

// ---------------------------------------------------------------------------
// host-side orchestration
// ---------------------------------------------------------------------------
extern "C" void kernel_launch(void* const* d_in, const int* in_sizes, int n_in,
                              void* d_out, int out_size, void* d_ws,
                              size_t ws_size, hipStream_t stream) {
  (void)in_sizes; (void)n_in; (void)out_size; (void)ws_size;

  const float* x   = (const float*)d_in[0];
  const float* Wg  = (const float*)d_in[1];
  const float* Weg = (const float*)d_in[2];
  const float* Weu = (const float*)d_in[3];
  const float* Wed = (const float*)d_in[4];
  const float* Wsg = (const float*)d_in[5];
  const float* Wsu = (const float*)d_in[6];
  const float* Wsd = (const float*)d_in[7];
  float* Out = (float*)d_out;

  char* ws = (char*)d_ws;
  size_t off = 0;
  auto alloc = [&](size_t bytes) {
    off = (off + 255) & ~(size_t)255;
    char* p = ws + off;
    off += bytes;
    return p;
  };

  __bf16* xb  = (__bf16*)alloc((size_t)T_TOK * DH * 2);
  __bf16* wsg = (__bf16*)alloc((size_t)DH * SDIM * 2);
  __bf16* wsu = (__bf16*)alloc((size_t)DH * SDIM * 2);
  __bf16* wsd = (__bf16*)alloc((size_t)SDIM * DH * 2);
  __bf16* weg = (__bf16*)alloc((size_t)NEXP * DH * DE * 2);
  __bf16* weu = (__bf16*)alloc((size_t)NEXP * DH * DE * 2);
  __bf16* wed = (__bf16*)alloc((size_t)NEXP * DE * DH * 2);
  __bf16* hs  = (__bf16*)alloc((size_t)T_TOK * SDIM * 2);
  __bf16* he  = (__bf16*)alloc(((size_t)2 * T_TOK + 128) * DE * 2);
  int*   counts = (int*)alloc(NEXP * 4);
  int*   offs   = (int*)alloc((NEXP + 1) * 4);
  int*   idxl   = (int*)alloc((size_t)NEXP * T_TOK * 4);
  float* wgtl   = (float*)alloc((size_t)NEXP * T_TOK * 4);

  // 1) fp32 -> bf16 conversions
  k_cvt<<<2048, 256, 0, stream>>>(x,   xb,  (long)T_TOK * DH);
  k_cvt<<<1024, 256, 0, stream>>>(Wsg, wsg, (long)DH * SDIM);
  k_cvt<<<1024, 256, 0, stream>>>(Wsu, wsu, (long)DH * SDIM);
  k_cvt<<<1024, 256, 0, stream>>>(Wsd, wsd, (long)SDIM * DH);
  k_cvt<<<2048, 256, 0, stream>>>(Weg, weg, (long)NEXP * DH * DE);
  k_cvt<<<2048, 256, 0, stream>>>(Weu, weu, (long)NEXP * DH * DE);
  k_cvt<<<2048, 256, 0, stream>>>(Wed, wed, (long)NEXP * DE * DH);

  // 2) routing
  k_zero_counts<<<1, 32, 0, stream>>>(counts);
  k_router<<<T_TOK / 8, 256, 0, stream>>>(x, Wg, counts, idxl, wgtl);
  k_scan<<<1, 1, 0, stream>>>(counts, offs);

  // 3) shared expert: hs = silu(x@Wsg)*(x@Wsu); Out = hs@Wsd
  k_swiglu<<<dim3(T_TOK / 128, SDIM / 64, 1), 256, 0, stream>>>(
      xb, wsg, wsu, (size_t)0, SDIM, hs, nullptr, nullptr, nullptr);
  k_down<<<dim3(T_TOK / 128, DH / 128, 1), 256, 0, stream>>>(
      hs, wsd, (size_t)0, SDIM, Out, nullptr, nullptr, nullptr, nullptr);

  // 4) routed experts (gathered rows, packed hidden buffer, scaled atomic add)
  k_swiglu<<<dim3(T_TOK / 128, DE / 64, NEXP), 256, 0, stream>>>(
      xb, weg, weu, (size_t)DH * DE, DE, he, counts, offs, idxl);
  k_down<<<dim3(T_TOK / 128, DH / 128, NEXP), 256, 0, stream>>>(
      he, wed, (size_t)DE * DH, DE, Out, counts, offs, idxl, wgtl);
}
